// Fuzzy_MultiHeadAttention_PA_69200513073603
// MI455X (gfx1250) — matrix operand order repaired and checked by reading, not compiled, not run on hardware
//
#include <hip/hip_runtime.h>
#include <hip/hip_bf16.h>

// ---------------- problem constants ----------------
constexpr int BB = 2;
constexpr int SS = 1024;
constexpr int EE = 768;
constexpr int HH = 12;
constexpr int RR = 32;
constexpr int DD = 64;          // head dim
constexpr int RS = RR + SS;     // 1056
constexpr float QSCALE = 0.125f;   // D^-0.5
constexpr float ESCALE = 8.0f;     // energy / scale = energy * sqrt(D)

typedef __bf16 bf16_t;
typedef __attribute__((ext_vector_type(16))) __bf16 v16bf;
typedef __attribute__((ext_vector_type(8)))  float  v8f;
typedef __attribute__((ext_vector_type(4)))  int    i4v;

union Frag { v16bf v; i4v q[2]; };

#define DEV __device__ __forceinline__

// A fragment (16x32 bf16, M x K). src: row-major [row][col], ld elements/row.
// ISA layout: lanes 0-15 row m, VGPR0-3 = K[half*8 .. +7], VGPR4-7 = K[16+half*8 .. +7].
DEV v16bf load_a_frag(const bf16_t* __restrict__ base, int ld, int col0, int lane) {
  int half = (lane >> 4) & 1;
  int m    = lane & 15;
  const bf16_t* row = base + (size_t)m * ld;
  Frag f;
  f.q[0] = *(const i4v*)(row + col0 + half * 8);
  f.q[1] = *(const i4v*)(row + col0 + 16 + half * 8);
  return f.v;
}

// B fragment (32x16 bf16, K x N). srcT: B^T row-major, i.e. element (k,n) at
// srcT[(n0+n)*ldt + k0 + k].  Lane (khalf*16+n) holds K[khalf*8 .. +7], K[16+khalf*8 .. +7].
DEV v16bf load_b_frag(const bf16_t* __restrict__ srcT, int ldt, int n0, int k0, int lane) {
  int khalf = (lane >> 4) & 1;
  int n     = lane & 15;
  const bf16_t* row = srcT + (size_t)(n0 + n) * ldt;
  Frag f;
  f.q[0] = *(const i4v*)(row + k0 + khalf * 8);
  f.q[1] = *(const i4v*)(row + k0 + 16 + khalf * 8);
  return f.v;
}

DEV v8f wmma_bf16(v16bf a, v16bf b, v8f c) {
  return __builtin_amdgcn_wmma_f32_16x16x32_bf16(false, a, false, b, (short)0, c,
                                                 false, false);
}

DEV v8f zero8() { v8f z = {0.f,0.f,0.f,0.f,0.f,0.f,0.f,0.f}; return z; }

// ---------------- conversion kernels ----------------
__global__ void cvt_f32_bf16(const float* __restrict__ s, bf16_t* __restrict__ d, int n) {
  int i = blockIdx.x * blockDim.x + threadIdx.x;
  if (i < n) d[i] = (bf16_t)s[i];
}

// value [B,S,E] f32 -> valT [B,H,D,S] bf16  (valT[b,h,d,s] = value[b,s,h*D+d])
__global__ void cvt_valT(const float* __restrict__ value, bf16_t* __restrict__ valT) {
  int i = blockIdx.x * blockDim.x + threadIdx.x;
  int total = BB * HH * DD * SS;
  if (i >= total) return;
  int s = i % SS;
  int d = (i / SS) % DD;
  int h = (i / (SS * DD)) % HH;
  int b = i / (SS * DD * HH);
  valT[i] = (bf16_t)value[((size_t)(b * SS + s)) * EE + h * DD + d];
}

// ---------------- per-head q/k projection: y = x @ W^T ----------------
// X: [B,S,E] bf16 (token rows), W: [D,D] bf16 row-major (= B^T for WMMA).
// out: [B,H,S,D] bf16 (+ compile-time optional f32 copy for the fuzzy branch).
template <bool WITH_F32>
__global__ __launch_bounds__(32) void proj_head64(const bf16_t* __restrict__ X,
                                                  const bf16_t* __restrict__ W,
                                                  bf16_t* __restrict__ out_bf,
                                                  float* __restrict__ out_f) {
  int lane  = threadIdx.x & 31;
  int bid   = blockIdx.x;
  int stile = bid % (SS / 16);
  int h     = (bid / (SS / 16)) % HH;
  int b     = bid / ((SS / 16) * HH);

  const bf16_t* Abase = X + ((size_t)(b * SS + stile * 16)) * EE + h * DD;
  v16bf a0 = load_a_frag(Abase, EE, 0, lane);
  v16bf a1 = load_a_frag(Abase, EE, 32, lane);

  int half = lane >> 4, n = lane & 15;
#pragma unroll
  for (int nt = 0; nt < 4; ++nt) {
    v16bf b0 = load_b_frag(W, DD, nt * 16, 0, lane);
    v16bf b1 = load_b_frag(W, DD, nt * 16, 32, lane);
    v8f acc = zero8();
    acc = wmma_bf16(a0, b0, acc);
    acc = wmma_bf16(a1, b1, acc);
#pragma unroll
    for (int r = 0; r < 8; ++r) {
      int m = r + 8 * half;
      size_t idx = (((size_t)(b * HH + h)) * SS + stile * 16 + m) * DD + nt * 16 + n;
      out_bf[idx] = (bf16_t)acc[r];
      if constexpr (WITH_F32) out_f[idx] = acc[r];
    }
  }
}

// ---------------- value projection: v[b,h,d,r] = sum_s Wv[r,s]*val[b,s,h,d] + bv[r] -----
// C[1056,64] = Wv[1056,1024] x Val[1024,64] per (b,h); stored transposed as [B,H,D,RS] bf16.
// Software-pipelined, unrolled x2 so the two fragment buffers alternate roles
// (loads land directly in loop-carried registers; no rotation copies).
__global__ __launch_bounds__(32) void vproj_kernel(const bf16_t* __restrict__ Wvb,
                                                   const bf16_t* __restrict__ valT,
                                                   const float* __restrict__ bv,
                                                   bf16_t* __restrict__ vout) {
  constexpr int MT = RS / 16;  // 66
  constexpr int KS = SS / 32;  // 32 (even)
  int lane = threadIdx.x & 31;
  int bid  = blockIdx.x;
  int mt = bid % MT;
  int h  = (bid / MT) % HH;
  int b  = bid / (MT * HH);

  const bf16_t* Abase = Wvb + (size_t)(mt * 16) * SS;                 // ld = S
  const bf16_t* Bbase = valT + ((size_t)(b * HH + h)) * DD * SS;      // [d][s], ldt = S

  v8f acc[4];
#pragma unroll
  for (int nt = 0; nt < 4; ++nt) acc[nt] = zero8();

  v16bf a0 = load_a_frag(Abase, SS, 0, lane);
  v16bf b0[4];
#pragma unroll
  for (int nt = 0; nt < 4; ++nt) b0[nt] = load_b_frag(Bbase, SS, nt * 16, 0, lane);

  for (int ks = 0; ks < KS; ks += 2) {
    // prefetch stage for ks+1 into buffer 1
    v16bf a1 = load_a_frag(Abase, SS, (ks + 1) * 32, lane);
    v16bf b1[4];
#pragma unroll
    for (int nt = 0; nt < 4; ++nt)
      b1[nt] = load_b_frag(Bbase, SS, nt * 16, (ks + 1) * 32, lane);
    // consume buffer 0 (step ks)
#pragma unroll
    for (int nt = 0; nt < 4; ++nt) acc[nt] = wmma_bf16(a0, b0[nt], acc[nt]);
    // refill buffer 0 with step ks+2 (skipped only on the final trip)
    if (ks + 2 < KS) {
      a0 = load_a_frag(Abase, SS, (ks + 2) * 32, lane);
#pragma unroll
      for (int nt = 0; nt < 4; ++nt)
        b0[nt] = load_b_frag(Bbase, SS, nt * 16, (ks + 2) * 32, lane);
    }
    // consume buffer 1 (step ks+1)
#pragma unroll
    for (int nt = 0; nt < 4; ++nt) acc[nt] = wmma_bf16(a1, b1[nt], acc[nt]);
  }

  int half = lane >> 4, n = lane & 15;
#pragma unroll
  for (int nt = 0; nt < 4; ++nt) {
#pragma unroll
    for (int r = 0; r < 8; ++r) {
      int rrow = mt * 16 + r + 8 * half;
      int d    = nt * 16 + n;
      float val = acc[nt][r] + bv[rrow];
      vout[(((size_t)(b * HH + h)) * DD + d) * RS + rrow] = (bf16_t)val;
    }
  }
}

// ---------------- fused fuzzy + flash attention ----------------
// One wave per 16 query rows of one (b,h). Output att [B,S,E] bf16.
__global__ __launch_bounds__(32) void flash_kernel(const bf16_t* __restrict__ qbf,
                                                   const float* __restrict__ qf,
                                                   const bf16_t* __restrict__ kbf,
                                                   const bf16_t* __restrict__ vbf,
                                                   const float* __restrict__ rk,
                                                   const float* __restrict__ rw,
                                                   bf16_t* __restrict__ att) {
  __shared__ __align__(16) bf16_t Pbuf[16 * 32];

  int lane  = threadIdx.x & 31;
  int bid   = blockIdx.x;
  int stile = bid % (SS / 16);
  int h     = (bid / (SS / 16)) % HH;
  int b     = bid / ((SS / 16) * HH);
  size_t bh = (size_t)(b * HH + h);

  const bf16_t* qbase  = qbf + (bh * SS + stile * 16) * DD;
  const float*  qfbase = qf  + (bh * SS + stile * 16) * DD;
  const bf16_t* kbase  = kbf + bh * SS * DD;        // [j][d], ldt = D  (= B^T)
  const bf16_t* vbase  = vbf + bh * DD * RS;        // [d][r], ldt = RS (= B^T)

  v16bf qa0 = load_a_frag(qbase, DD, 0, lane);
  v16bf qa1 = load_a_frag(qbase, DD, 32, lane);

  // ---- fuzzy branch: z[s,r] = -0.5/D * sum_d ((q*0.125 - rk)/rw)^2 ; softmax over R ----
  {
    int fm   = lane & 15;
    int rblk = (lane >> 4) * 16;
    const float* qp = qfbase + fm * DD;
    float p[16];
    float zmax = -1e30f;
#pragma unroll
    for (int rr = 0; rr < 16; ++rr) {
      int r = rblk + rr;
      const float* rkp = rk + ((size_t)h * RR + r) * DD;
      const float* rwp = rw + ((size_t)h * RR + r) * DD;
      float acc = 0.f;
      for (int d = 0; d < DD; ++d) {
        float t = (qp[d] * QSCALE - rkp[d]) / rwp[d];
        acc = fmaf(t, t, acc);
      }
      p[rr] = acc * (-0.5f / (float)DD);
      zmax = fmaxf(zmax, p[rr]);
    }
    zmax = fmaxf(zmax, __shfl_xor(zmax, 16));
    float sum = 0.f;
#pragma unroll
    for (int rr = 0; rr < 16; ++rr) { p[rr] = __expf(p[rr] - zmax); sum += p[rr]; }
    sum += __shfl_xor(sum, 16);
    float inv = 1.f / sum;
#pragma unroll
    for (int rr = 0; rr < 16; ++rr)
      Pbuf[fm * 32 + rblk + rr] = (bf16_t)(p[rr] * inv);
  }
  __syncthreads();

  v8f ofz[4];
  {
    v16bf pa = load_a_frag(Pbuf, 32, 0, lane);
#pragma unroll
    for (int nt = 0; nt < 4; ++nt) {
      v16bf vb = load_b_frag(vbase, RS, nt * 16, 0, lane);   // r = 0..31 (fuzzy slots)
      ofz[nt] = wmma_bf16(pa, vb, zero8());
    }
  }
  __syncthreads();

  // ---- dot branch, flash style over 32 keys per iteration ----
  v8f o[4];
#pragma unroll
  for (int nt = 0; nt < 4; ++nt) o[nt] = zero8();
  float mrow[8], lrow[8];
#pragma unroll
  for (int r = 0; r < 8; ++r) { mrow[r] = -1e30f; lrow[r] = 0.f; }

  int half = lane >> 4, nn = lane & 15;

  for (int jt = 0; jt < SS / 32; ++jt) {
    const bf16_t* kb = kbase + (size_t)(jt * 32) * DD;

    // prefetch next iteration's K tile (global_prefetch_b8; one byte per lane
    // pulls a full cacheline, 32 lanes cover the 32x64 bf16 tile)
    if (jt + 1 < SS / 32) {
      __builtin_prefetch(kb + (size_t)32 * DD + (size_t)lane * DD, 0, 3);
    }

    v8f s0 = zero8(), s1 = zero8();
    {
      v16bf b00 = load_b_frag(kb, DD, 0, 0, lane);
      v16bf b01 = load_b_frag(kb, DD, 0, 32, lane);
      s0 = wmma_bf16(qa0, b00, s0);
      s0 = wmma_bf16(qa1, b01, s0);
      v16bf b10 = load_b_frag(kb, DD, 16, 0, lane);
      v16bf b11 = load_b_frag(kb, DD, 16, 32, lane);
      s1 = wmma_bf16(qa0, b10, s1);
      s1 = wmma_bf16(qa1, b11, s1);
    }

    // issue the PV value-fragment loads now so they are in flight
    // behind the scalar softmax work below
    v16bf vbfrag[4];
#pragma unroll
    for (int nt = 0; nt < 4; ++nt)
      vbfrag[nt] = load_b_frag(vbase, RS, nt * 16, RR + jt * 32, lane);

    float alpha[8];
#pragma unroll
    for (int r = 0; r < 8; ++r) {
      float e0 = s0[r] * ESCALE;
      float e1 = s1[r] * ESCALE;
      float v = fmaxf(e0, e1);
      v = fmaxf(v, __shfl_xor(v, 1, 16));
      v = fmaxf(v, __shfl_xor(v, 2, 16));
      v = fmaxf(v, __shfl_xor(v, 4, 16));
      v = fmaxf(v, __shfl_xor(v, 8, 16));
      float mnew = fmaxf(mrow[r], v);
      float a = __expf(mrow[r] - mnew);
      mrow[r] = mnew;
      float p0 = __expf(e0 - mnew);
      float p1 = __expf(e1 - mnew);
      float ssum = p0 + p1;
      ssum += __shfl_xor(ssum, 1, 16);
      ssum += __shfl_xor(ssum, 2, 16);
      ssum += __shfl_xor(ssum, 4, 16);
      ssum += __shfl_xor(ssum, 8, 16);
      lrow[r] = lrow[r] * a + ssum;
      alpha[r] = a;
      Pbuf[(r + 8 * half) * 32 + nn]      = (bf16_t)p0;
      Pbuf[(r + 8 * half) * 32 + 16 + nn] = (bf16_t)p1;
    }
#pragma unroll
    for (int nt = 0; nt < 4; ++nt)
#pragma unroll
      for (int r = 0; r < 8; ++r) o[nt][r] *= alpha[r];

    __syncthreads();
    v16bf pa = load_a_frag(Pbuf, 32, 0, lane);
#pragma unroll
    for (int nt = 0; nt < 4; ++nt) o[nt] = wmma_bf16(pa, vbfrag[nt], o[nt]);
    __syncthreads();
  }

  // ---- epilogue: out = O_dot / l + O_fuzzy -> att[b, s, h*D + d] (bf16) ----
#pragma unroll
  for (int nt = 0; nt < 4; ++nt) {
#pragma unroll
    for (int r = 0; r < 8; ++r) {
      int m = r + 8 * half;
      float val = o[nt][r] / lrow[r] + ofz[nt][r];
      size_t idx = ((size_t)(b * SS) + stile * 16 + m) * EE + h * DD + nt * 16 + nn;
      att[idx] = (bf16_t)val;
    }
  }
}

// ---------------- final projection: out = att @ Wo^T + bo ----------------
// Software-pipelined, unrolled x2 (same buffer-alternation scheme as vproj).
__global__ __launch_bounds__(32) void final_proj(const bf16_t* __restrict__ att,
                                                 const bf16_t* __restrict__ Wob,
                                                 const float* __restrict__ bo,
                                                 float* __restrict__ out) {
  constexpr int NG = EE / 64;  // 12 groups of 64 output cols
  constexpr int KS = EE / 32;  // 24 (even)
  int lane = threadIdx.x & 31;
  int bid  = blockIdx.x;
  int ng = bid % NG;
  int mt = bid / NG;           // 0 .. B*S/16-1

  const bf16_t* Abase = att + (size_t)(mt * 16) * EE;

  v8f acc[4];
#pragma unroll
  for (int nt = 0; nt < 4; ++nt) acc[nt] = zero8();

  v16bf a0 = load_a_frag(Abase, EE, 0, lane);
  v16bf b0[4];
#pragma unroll
  for (int nt = 0; nt < 4; ++nt)
    b0[nt] = load_b_frag(Wob, EE, ng * 64 + nt * 16, 0, lane);

  for (int ks = 0; ks < KS; ks += 2) {
    v16bf a1 = load_a_frag(Abase, EE, (ks + 1) * 32, lane);
    v16bf b1[4];
#pragma unroll
    for (int nt = 0; nt < 4; ++nt)
      b1[nt] = load_b_frag(Wob, EE, ng * 64 + nt * 16, (ks + 1) * 32, lane);
#pragma unroll
    for (int nt = 0; nt < 4; ++nt) acc[nt] = wmma_bf16(a0, b0[nt], acc[nt]);
    if (ks + 2 < KS) {
      a0 = load_a_frag(Abase, EE, (ks + 2) * 32, lane);
#pragma unroll
      for (int nt = 0; nt < 4; ++nt)
        b0[nt] = load_b_frag(Wob, EE, ng * 64 + nt * 16, (ks + 2) * 32, lane);
    }
#pragma unroll
    for (int nt = 0; nt < 4; ++nt) acc[nt] = wmma_bf16(a1, b1[nt], acc[nt]);
  }

  int half = lane >> 4, n = lane & 15;
#pragma unroll
  for (int nt = 0; nt < 4; ++nt) {
    int ocol = ng * 64 + nt * 16 + n;
    float bias = bo[ocol];
#pragma unroll
    for (int r = 0; r < 8; ++r) {
      int m = r + 8 * half;
      out[((size_t)(mt * 16) + m) * EE + ocol] = acc[nt][r] + bias;
    }
  }
}

// ---------------- host launch ----------------
extern "C" void kernel_launch(void* const* d_in, const int* in_sizes, int n_in,
                              void* d_out, int out_size, void* d_ws, size_t ws_size,
                              hipStream_t stream) {
  const float* query = (const float*)d_in[0];
  const float* key   = (const float*)d_in[1];
  const float* value = (const float*)d_in[2];
  const float* rk    = (const float*)d_in[3];
  const float* rw    = (const float*)d_in[4];
  const float* Wq    = (const float*)d_in[5];
  const float* Wk    = (const float*)d_in[6];
  const float* Wv    = (const float*)d_in[7];
  const float* bv    = (const float*)d_in[8];
  const float* Wo    = (const float*)d_in[9];
  const float* bo    = (const float*)d_in[10];
  float* out = (float*)d_out;

  char* w = (char*)d_ws;
  auto alloc = [&](size_t elems) {
    void* p = (void*)w;
    w += (elems * sizeof(bf16_t) + 255) & ~(size_t)255;
    return (bf16_t*)p;
  };

  const int nTok = BB * SS * EE;          // 1,572,864
  bf16_t* qx_bf  = alloc(nTok);
  bf16_t* kx_bf  = alloc(nTok);
  bf16_t* valT   = alloc((size_t)BB * HH * DD * SS);
  bf16_t* Wq_bf  = alloc(DD * DD);
  bf16_t* Wk_bf  = alloc(DD * DD);
  bf16_t* Wv_bf  = alloc((size_t)RS * SS);
  bf16_t* Wo_bf  = alloc((size_t)EE * EE);
  bf16_t* q_bf   = alloc((size_t)BB * HH * SS * DD);
  bf16_t* k_bf   = alloc((size_t)BB * HH * SS * DD);
  bf16_t* v_bf   = alloc((size_t)BB * HH * DD * RS);
  bf16_t* att_bf = alloc(nTok);
  float* q_f = (float*)w;
  w += ((size_t)BB * HH * SS * DD * sizeof(float) + 255) & ~(size_t)255;

  auto cvt = [&](const float* s, bf16_t* d, int n) {
    cvt_f32_bf16<<<(n + 255) / 256, 256, 0, stream>>>(s, d, n);
  };
  cvt(query, qx_bf, nTok);
  cvt(key,   kx_bf, nTok);
  cvt(Wq, Wq_bf, DD * DD);
  cvt(Wk, Wk_bf, DD * DD);
  cvt(Wv, Wv_bf, RS * SS);
  cvt(Wo, Wo_bf, EE * EE);
  {
    int n = BB * HH * DD * SS;
    cvt_valT<<<(n + 255) / 256, 256, 0, stream>>>(value, valT);
  }

  int projBlocks = BB * HH * (SS / 16);               // 1536
  proj_head64<true ><<<projBlocks, 32, 0, stream>>>(qx_bf, Wq_bf, q_bf, q_f);
  proj_head64<false><<<projBlocks, 32, 0, stream>>>(kx_bf, Wk_bf, k_bf, nullptr);

  vproj_kernel<<<BB * HH * (RS / 16), 32, 0, stream>>>(Wv_bf, valT, bv, v_bf);

  flash_kernel<<<BB * HH * (SS / 16), 32, 0, stream>>>(q_bf, q_f, k_bf, v_bf,
                                                       rk, rw, att_bf);

  final_proj<<<(BB * SS / 16) * (EE / 64), 32, 0, stream>>>(att_bf, Wo_bf, bo, out);
}
